// DilatedSparseRnnStack_71262097375516
// MI455X (gfx1250) — matrix operand order repaired and checked by reading, compiled, not run on hardware
//
#include <hip/hip_runtime.h>

// DilatedSparseRnnStack for MI455X (gfx1250, wave32, WMMA).
//
// Strategy: the recurrence is independent per batch row, so batch is
// partitioned across 16 workgroups (16 rows each = one WMMA M-tile).
// Each WG runs the whole T=256 scan of a layer with weights (f16),
// c-history (f32) and h-history (f16) resident in 320KB LDS; the per-step
// [16,192]@[192,512] GEMM is 192 v_wmma_f32_16x16x32_f16 spread over 8 waves.
// Wave w owns gate column tiles {w, w+8, w+16, w+24} so the f/ns/alpha/o
// gates of state columns 16w..16w+15 land in that wave's accumulators and
// the gating runs in registers. Inter-layer activations live in d_ws
// (requires ~33.6 MB; stays in the 192MB L2).

typedef _Float16 half8  __attribute__((ext_vector_type(8)));
typedef _Float16 half16 __attribute__((ext_vector_type(16)));
typedef float    float8 __attribute__((ext_vector_type(8)));

__device__ __forceinline__ half16 ldfrag(const _Float16* p, int off2) {
  half8 lo = *(const half8*)p;
  half8 hi = *(const half8*)(p + off2);
  return __builtin_shufflevector(lo, hi, 0,1,2,3,4,5,6,7,8,9,10,11,12,13,14,15);
}

__device__ __forceinline__ float fast_sig(float x) {
  return __builtin_amdgcn_rcpf(1.0f + __expf(-x));
}
__device__ __forceinline__ float fast_tanh(float x) {
  return 2.0f * fast_sig(2.0f * x) - 1.0f;
}

// ---------------------------------------------------------------------------
// Sequential scan of one S2 layer for a 16-row batch slice.
// LDS layout (dynamic, 325632 B total):
//   [0      ,196608) : sW   f16 [512][192]   layer weights
//   [196608 ,294912) : sC   f32 [12][16][128] c circular buffer
//   [294912 ,301056) : sXH  f16 [16][192]    {x | prevH | dH} GEMM input
//   [301056 ,325632) : sHb  f16 [12][16][64] h circular buffer
// ---------------------------------------------------------------------------
__global__ __launch_bounds__(256)
void s2_scan_kernel(const float* __restrict__ in, float* __restrict__ out,
                    const float* __restrict__ Wg, const float* __restrict__ bv,
                    int d, const float* __restrict__ res, int addRes)
{
  extern __shared__ char smem[];
  _Float16* sW  = (_Float16*)(smem);
  float*    sC  = (float*)   (smem + 196608);
  _Float16* sXH = (_Float16*)(smem + 294912);
  _Float16* sHb = (_Float16*)(smem + 301056);

  const int tid  = threadIdx.x;
  const int b0   = blockIdx.x * 16;          // batch slice base row
  const int w    = tid >> 5;                 // wave id 0..7
  const int lane = tid & 31;
  const int nloc = lane & 15;                // N within tile
  const int hv   = lane >> 4;                // lane-half (row group / k group)
  const int nst  = w * 16 + nloc;            // owned state column 0..127

  // Load layer weights f32 -> f16 into LDS (coalesced).
  #pragma unroll 8
  for (int j = 0; j < 384; ++j) {
    int i = j * 256 + tid;
    sW[i] = (_Float16)Wg[i];
  }

  // Per-lane bias values for the 4 gates of this state column.
  const float bs0 = bv[nst] + 1.0f;          // f-gate bias includes the +1
  const float bs1 = bv[128 + nst];
  const float bs2 = bv[256 + nst];
  const float bs3 = bv[384 + nst];

  float cPrev[8];
  #pragma unroll
  for (int r = 0; r < 8; ++r) cPrev[r] = 0.0f;

  __syncthreads();

  int slot = 0;                               // t % d, maintained incrementally
  for (int t = 0; t < 256; ++t) {
    // ---- Phase A: stage GEMM input xh = {x_t | prevH | dH} as f16 --------
    #pragma unroll
    for (int j = 0; j < 4; ++j) {             // x part: 16x64
      int i = j * 256 + tid;
      int m = i >> 6, c = i & 63;
      sXH[m * 192 + c] = (_Float16)in[((size_t)t * 256 + b0 + m) * 64 + c];
    }
    if (t == 0) {                             // prevH = dH = 0
      #pragma unroll
      for (int j = 0; j < 8; ++j) {
        int i = j * 256 + tid;
        int m = i >> 7, c = i & 127;
        sXH[m * 192 + 64 + c] = (_Float16)0.0f;
      }
    } else if (t < d) {                       // dH := prevH
      #pragma unroll
      for (int j = 0; j < 4; ++j) {
        int i = j * 256 + tid;
        int m = i >> 6, c = i & 63;
        sXH[m * 192 + 128 + c] = sXH[m * 192 + 64 + c];
      }
    } else {                                  // dH := h from step t-d
      #pragma unroll
      for (int j = 0; j < 4; ++j) {
        int i = j * 256 + tid;
        int m = i >> 6, c = i & 63;
        sXH[m * 192 + 128 + c] = sHb[slot * 1024 + m * 64 + c];
      }
    }
    __syncthreads();

    // ---- Phase B: gates[16,512] = xh[16,192] @ W^T via WMMA --------------
    float8 acc[4] = {};
    half16 aF[6];
    #pragma unroll
    for (int kt = 0; kt < 6; ++kt)            // A frags: row nloc, K pattern
      aF[kt] = ldfrag(&sXH[nloc * 192 + kt * 32 + hv * 8], 16);

    #pragma unroll
    for (int g = 0; g < 4; ++g) {             // gate g for owned state cols
      const int row = (w + 8 * g) * 16 + nloc;  // gate (output) column
      #pragma unroll
      for (int kt = 0; kt < 6; ++kt) {
        half16 bF = ldfrag(&sW[row * 192 + kt * 32 + hv * 8], 16);
        acc[g] = __builtin_amdgcn_wmma_f32_16x16x32_f16(
            false, aF[kt], false, bF, (short)0, acc[g], false, false);
      }
    }
    __syncthreads();   // all A reads done before h columns are overwritten

    // ---- Phase C: gating in accumulator registers ------------------------
    #pragma unroll
    for (int r = 0; r < 8; ++r) {
      const int m = hv * 8 + r;               // batch row in slice
      float f  = fast_sig (acc[0][r] + bs0);
      float ns = fast_tanh(acc[1][r] + bs1);
      float al = fast_sig (acc[2][r] + bs2);
      float o  = fast_sig (acc[3][r] + bs3);

      float cp = cPrev[r], wc;
      if (t >= d) {
        float dC = sC[slot * 2048 + m * 128 + nst];  // written by this lane at t-d
        wc = al * cp + (1.0f - al) * dC;
      } else {
        wc = cp;
      }
      float nc = (t > 0) ? (f * wc + (1.0f - f) * ns) : ns;
      float wh = o * nc;

      cPrev[r] = nc;
      sC[slot * 2048 + m * 128 + nst] = nc;   // read-before-write slot reuse

      const size_t oi = ((size_t)t * 256 + b0 + m) * 64;
      if (nst < 64) {                         // "out" half of the state
        float v = wh;
        if (addRes) v += res[oi + nst];       // fused block residual
        out[oi + nst] = v;
      } else {                                // "h" half: feed recurrence
        _Float16 hh = (_Float16)wh;
        sXH[m * 192 + nst] = hh;              // prevH slot (cols 64..127)
        sHb[slot * 1024 + m * 64 + (nst - 64)] = hh;
      }
    }
    __syncthreads();
    slot = (slot + 1 == d) ? 0 : slot + 1;
  }
}

// ---------------------------------------------------------------------------
// Final projection: Y[65536,64] = A[65536,64] @ Wa^T + ba  (WMMA, parallel)
// ---------------------------------------------------------------------------
__global__ __launch_bounds__(256)
void s2_out_gemm(const float* __restrict__ A, const float* __restrict__ Wa,
                 const float* __restrict__ ba, float* __restrict__ Y)
{
  __shared__ _Float16 sA [128 * 64];
  __shared__ _Float16 sWa[64 * 64];

  const int tid = threadIdx.x;
  const int blockRow0 = blockIdx.x * 128;

  #pragma unroll
  for (int j = 0; j < 16; ++j) {
    int i = j * 256 + tid;
    sWa[i] = (_Float16)Wa[i];
  }
  #pragma unroll
  for (int j = 0; j < 32; ++j) {
    int i = j * 256 + tid;
    sA[i] = (_Float16)A[(size_t)blockRow0 * 64 + i];
  }
  __syncthreads();

  const int w = tid >> 5, lane = tid & 31, nloc = lane & 15, hv = lane >> 4;
  const int mrow = w * 16 + nloc;             // A row within staged block

  half16 aF[2];
  #pragma unroll
  for (int kt = 0; kt < 2; ++kt)
    aF[kt] = ldfrag(&sA[mrow * 64 + kt * 32 + hv * 8], 16);

  float8 acc[4] = {};
  #pragma unroll
  for (int g = 0; g < 4; ++g) {
    const int row = g * 16 + nloc;            // output column i
    #pragma unroll
    for (int kt = 0; kt < 2; ++kt) {
      half16 bF = ldfrag(&sWa[row * 64 + kt * 32 + hv * 8], 16);
      acc[g] = __builtin_amdgcn_wmma_f32_16x16x32_f16(
          false, aF[kt], false, bF, (short)0, acc[g], false, false);
    }
  }

  #pragma unroll
  for (int g = 0; g < 4; ++g) {
    const int i = g * 16 + nloc;
    const float bai = ba[i];
    #pragma unroll
    for (int r = 0; r < 8; ++r) {
      const int m = hv * 8 + r;
      Y[((size_t)blockRow0 + w * 16 + m) * 64 + i] = acc[g][r] + bai;
    }
  }
}

// ---------------------------------------------------------------------------
extern "C" void kernel_launch(void* const* d_in, const int* in_sizes, int n_in,
                              void* d_out, int out_size, void* d_ws, size_t ws_size,
                              hipStream_t stream) {
  (void)in_sizes; (void)n_in; (void)out_size; (void)ws_size;

  const float* x  = (const float*)d_in[0];   // [256,256,64]
  const float* W  = (const float*)d_in[1];   // [4,512,192]
  const float* bb = (const float*)d_in[2];   // [4,512]
  const float* Wa = (const float*)d_in[3];   // [64,64]
  const float* ba = (const float*)d_in[4];   // [64]
  float* Yout = (float*)d_out;               // [256,256,64] f32

  // Scratch: two T*B*64 f32 activation buffers (2 * 16.78 MB).
  float* buf0 = (float*)d_ws;
  float* buf1 = buf0 + (size_t)256 * 256 * 64;

  const size_t SH = 325632;                  // dynamic LDS bytes (<= 320KB)
  const size_t WS = (size_t)512 * 192;       // per-layer weight stride

  // Block 0: layers 0 (d=1) and 1 (d=3); res1 = L1out (+0 residual).
  s2_scan_kernel<<<16, 256, SH, stream>>>(x,    buf0, W + 0 * WS, bb + 0,    1,  nullptr, 0);
  s2_scan_kernel<<<16, 256, SH, stream>>>(buf0, buf1, W + 1 * WS, bb + 512,  3,  nullptr, 0);
  // Block 1: layers 2 (d=6) and 3 (d=12); res2 = L3out + res1 (fused add).
  s2_scan_kernel<<<16, 256, SH, stream>>>(buf1, buf0, W + 2 * WS, bb + 1024, 6,  nullptr, 0);
  s2_scan_kernel<<<16, 256, SH, stream>>>(buf0, buf1, W + 3 * WS, bb + 1536, 12, buf1,    1);

  // Final projection.
  s2_out_gemm<<<512, 256, 0, stream>>>(buf1, Wa, ba, Yout);
}
